// MultiHeadAttention_6055903887702
// MI455X (gfx1250) — matrix operand order repaired and checked by reading, not compiled, not run on hardware
//
#include <hip/hip_runtime.h>
#include <cstdint>
#include <cstddef>

// MHA: B=4, S=2048, E=768, H=8 (full-width heads), fp32 in/out.
// All matmuls run on v_wmma_f32_16x16x32_bf16 (bf16 in, fp32 accum).
// GEMM: 128x128 block tile, 8 waves (wave32), 32x64 wave tile,
//       double-buffered LDS with register-staged global prefetch.

#define BB 4
#define SS 2048
#define EE 768
#define HH 8

typedef __attribute__((ext_vector_type(16))) __bf16 v16bf;
typedef __attribute__((ext_vector_type(8)))  __bf16 v8bf;
typedef __attribute__((ext_vector_type(8)))  float  v8f;

static __device__ __forceinline__ __bf16 f2bf(float f) {
  union { float f; uint32_t u; } v; v.f = f;
  uint32_t r = v.u + 0x7FFFu + ((v.u >> 16) & 1u);   // round-to-nearest-even
  uint16_t h = (uint16_t)(r >> 16);
  return __builtin_bit_cast(__bf16, h);
}

__global__ void cvt_f32_bf16(const float* __restrict__ in,
                             __bf16* __restrict__ out, size_t n) {
  size_t i = (size_t)blockIdx.x * blockDim.x + threadIdx.x;
  size_t stride = (size_t)gridDim.x * blockDim.x;
  for (; i < n; i += stride) out[i] = f2bf(in[i]);
}

// ---------------------------------------------------------------------------
// Tiled WMMA GEMM: C[M,N] = A[M,K] * op(B)  (+ bias)
//   BT=false: B is [K,N] row-major.   BT=true: B is [N,K] row-major (B^T).
// ---------------------------------------------------------------------------
constexpr int BM = 128, BN = 128, BK = 32, BKP = 40;  // BKP: +8 pad, keeps 16B align

template <bool BT, bool OUT_BF16, bool BIAS>
__global__ __launch_bounds__(256) void gemm_wmma_bf16(
    const __bf16* __restrict__ A, const __bf16* __restrict__ Bm,
    void* __restrict__ Cv, const float* __restrict__ bias,
    int K, int lda, int ldb, int ldc,
    size_t sAz, size_t sBz, size_t sCz) {
  A  += (size_t)blockIdx.z * sAz;
  Bm += (size_t)blockIdx.z * sBz;
  const size_t coff = (size_t)blockIdx.z * sCz;

  __shared__ __bf16 As[2][BM * BKP];   // [m][k] row-major
  __shared__ __bf16 Bs[2][BN * BKP];   // [n][k] (B transposed in LDS)

  const int t    = threadIdx.x;
  const int wave = t >> 5;
  const int lane = t & 31;
  const int lr   = lane & 15;          // row/col within 16
  const int lh   = lane >> 4;          // lane half selects K chunk
  const int wrow = (wave & 3) * 32;    // 4x2 wave grid, 32x64 wave tile
  const int wcol = (wave >> 2) * 64;
  const int blockRow = blockIdx.x * BM;
  const int blockCol = blockIdx.y * BN;

  v8f acc[2][4] = {};

  auto loadA = [&](int k0, v8bf (&ra)[2]) {
#pragma unroll
    for (int j = 0; j < 2; ++j) {
      int idx = t + j * 256;
      int r = idx >> 2, c = (idx & 3) * 8;
      ra[j] = *(const v8bf*)(A + (size_t)(blockRow + r) * lda + (k0 + c));
    }
  };
  auto loadB = [&](int k0, v8bf (&rb)[2]) {
    if (BT) {
#pragma unroll
      for (int j = 0; j < 2; ++j) {
        int idx = t + j * 256;
        int n = idx >> 2, c = (idx & 3) * 8;
        rb[j] = *(const v8bf*)(Bm + (size_t)(blockCol + n) * ldb + (k0 + c));
      }
    } else {
#pragma unroll
      for (int j = 0; j < 2; ++j) {
        int idx = t + j * 256;
        int kk = idx >> 4, n = (idx & 15) * 8;
        rb[j] = *(const v8bf*)(Bm + (size_t)(k0 + kk) * ldb + (blockCol + n));
      }
    }
  };
  auto storeLDS = [&](int buf, v8bf (&ra)[2], v8bf (&rb)[2]) {
#pragma unroll
    for (int j = 0; j < 2; ++j) {
      int idx = t + j * 256;
      int r = idx >> 2, c = (idx & 3) * 8;
      *(v8bf*)&As[buf][r * BKP + c] = ra[j];
    }
    if (BT) {
#pragma unroll
      for (int j = 0; j < 2; ++j) {
        int idx = t + j * 256;
        int n = idx >> 2, c = (idx & 3) * 8;
        *(v8bf*)&Bs[buf][n * BKP + c] = rb[j];
      }
    } else {
#pragma unroll
      for (int j = 0; j < 2; ++j) {
        int idx = t + j * 256;
        int kk = idx >> 4, n = (idx & 15) * 8;
#pragma unroll
        for (int e = 0; e < 8; ++e) Bs[buf][(n + e) * BKP + kk] = rb[j][e];
      }
    }
  };
  auto compute = [&](int buf) {
    v16bf af[2], bfr[4];
#pragma unroll
    for (int s2 = 0; s2 < 2; ++s2) {      // A frags per ISA wave32 layout
      int r = wrow + s2 * 16 + lr;
      v8bf lo = *(const v8bf*)&As[buf][r * BKP + lh * 8];
      v8bf hi = *(const v8bf*)&As[buf][r * BKP + 16 + lh * 8];
#pragma unroll
      for (int e = 0; e < 8; ++e) { af[s2][e] = lo[e]; af[s2][e + 8] = hi[e]; }
    }
#pragma unroll
    for (int s2 = 0; s2 < 4; ++s2) {      // B frags: lane=col, K in same chunks as A
      int n = wcol + s2 * 16 + lr;
      v8bf lo = *(const v8bf*)&Bs[buf][n * BKP + lh * 8];
      v8bf hi = *(const v8bf*)&Bs[buf][n * BKP + 16 + lh * 8];
#pragma unroll
      for (int e = 0; e < 8; ++e) { bfr[s2][e] = lo[e]; bfr[s2][e + 8] = hi[e]; }
    }
#pragma unroll
    for (int i = 0; i < 2; ++i)
#pragma unroll
      for (int j = 0; j < 4; ++j)
        acc[i][j] = __builtin_amdgcn_wmma_f32_16x16x32_bf16(
            false, af[i], false, bfr[j], (short)0, acc[i][j], false, false);
  };

  // ---- software pipeline: prefetch(t+1) || compute(t), double-buffered LDS
  v8bf ra[2], rb[2];
  loadA(0, ra);
  loadB(0, rb);
  storeLDS(0, ra, rb);
  __syncthreads();

  int buf = 0;
  for (int k0 = 0; k0 < K; k0 += BK) {
    const int kn = k0 + BK;
    if (kn < K) {
      loadA(kn, ra);
      loadB(kn, rb);
      if (kn + BK < K) {   // warm L2/WGP$ two tiles ahead (global_prefetch_b8)
        __builtin_prefetch(A + (size_t)(blockRow + (t >> 2)) * lda + (kn + BK), 0, 1);
        __builtin_prefetch(Bm + (BT ? (size_t)(blockCol + (t >> 2)) * ldb + (kn + BK)
                                    : (size_t)(kn + BK + (t >> 4)) * ldb + blockCol),
                           0, 1);
      }
    }
    compute(buf);
    if (kn >= K) break;
    storeLDS(buf ^ 1, ra, rb);
    __syncthreads();
    buf ^= 1;
  }

  // ---- epilogue: C layout VGPR v -> M = v (lanes 0-15) / v+8 (lanes 16-31)
#pragma unroll
  for (int i = 0; i < 2; ++i) {
#pragma unroll
    for (int j = 0; j < 4; ++j) {
      int nn = blockCol + wcol + j * 16 + lr;
      float bv = BIAS ? bias[nn] : 0.0f;
#pragma unroll
      for (int v = 0; v < 8; ++v) {
        int mm = blockRow + wrow + i * 16 + (lh ? 8 : 0) + v;
        float val = acc[i][j][v] + bv;
        size_t o = coff + (size_t)mm * ldc + nn;
        if (OUT_BF16) ((__bf16*)Cv)[o] = f2bf(val);
        else          ((float*)Cv)[o]  = val;
      }
    }
  }
}

// ---------------------------------------------------------------------------
// Row softmax: P = softmax(att * scale) per row of 2048, fp32 in -> bf16 out.
// ---------------------------------------------------------------------------
__global__ __launch_bounds__(256) void softmax_rows(
    const float* __restrict__ att, __bf16* __restrict__ P, float scale) {
  const int row = blockIdx.x;
  const float* in = att + (size_t)row * SS;
  __bf16* out = P + (size_t)row * SS;
  const int t = threadIdx.x, lane = t & 31, wave = t >> 5;

  float vals[8];
  float m = -3.402823466e+38f;
#pragma unroll
  for (int i = 0; i < 8; ++i) {
    vals[i] = in[t + i * 256] * scale;
    m = fmaxf(m, vals[i]);
  }
#pragma unroll
  for (int off = 16; off; off >>= 1) m = fmaxf(m, __shfl_xor(m, off, 32));
  __shared__ float redm[8];
  if (lane == 0) redm[wave] = m;
  __syncthreads();
#pragma unroll
  for (int w = 0; w < 8; ++w) m = fmaxf(m, redm[w]);

  float s = 0.f;
#pragma unroll
  for (int i = 0; i < 8; ++i) { vals[i] = __expf(vals[i] - m); s += vals[i]; }
#pragma unroll
  for (int off = 16; off; off >>= 1) s += __shfl_xor(s, off, 32);
  __shared__ float reds[8];
  if (lane == 0) reds[wave] = s;
  __syncthreads();
  s = 0.f;
#pragma unroll
  for (int w = 0; w < 8; ++w) s += reds[w];
  float inv = 1.0f / s;
#pragma unroll
  for (int i = 0; i < 8; ++i) out[t + i * 256] = f2bf(vals[i] * inv);
}

// ---------------------------------------------------------------------------
extern "C" void kernel_launch(void* const* d_in, const int* in_sizes, int n_in,
                              void* d_out, int out_size, void* d_ws, size_t ws_size,
                              hipStream_t stream) {
  const float* x  = (const float*)d_in[0];
  const float* Wq = (const float*)d_in[1];
  const float* Wk = (const float*)d_in[2];
  const float* Wv = (const float*)d_in[3];
  const float* Wp = (const float*)d_in[4];
  const float* bp = (const float*)d_in[5];

  const size_t nx = (size_t)BB * SS * EE;        // 6.29M
  const size_t nW = (size_t)HH * EE * EE;        // 4.72M (also (H*E) x E for Wp)
  const size_t nQ = (size_t)HH * BB * SS * EE;   // 50.3M
  const size_t nAtt = (size_t)SS * SS;           // 4.19M

  char* ws = (char*)d_ws;
  size_t off = 0;
  __bf16* xb  = (__bf16*)(ws + off); off += nx * 2;
  __bf16* wqb = (__bf16*)(ws + off); off += nW * 2;
  __bf16* wkb = (__bf16*)(ws + off); off += nW * 2;
  __bf16* wvb = (__bf16*)(ws + off); off += nW * 2;
  __bf16* wpb = (__bf16*)(ws + off); off += nW * 2;
  __bf16* qb  = (__bf16*)(ws + off); off += nQ * 2;   // [H, B*S, E]
  __bf16* kb  = (__bf16*)(ws + off); off += nQ * 2;
  __bf16* vb  = (__bf16*)(ws + off); off += nQ * 2;
  __bf16* ob  = (__bf16*)(ws + off); off += nQ * 2;   // [B, S, H, E]
  float*  att = (float*) (ws + off); off += nAtt * 4; // per-(b,h) scores (L2-resident)
  __bf16* pb  = (__bf16*)(ws + off); off += nAtt * 2; // softmax probs

  // 1) fp32 -> bf16 conversions
  cvt_f32_bf16<<<2048, 256, 0, stream>>>(x,  xb,  nx);
  cvt_f32_bf16<<<2048, 256, 0, stream>>>(Wq, wqb, nW);
  cvt_f32_bf16<<<2048, 256, 0, stream>>>(Wk, wkb, nW);
  cvt_f32_bf16<<<2048, 256, 0, stream>>>(Wv, wvb, nW);
  cvt_f32_bf16<<<2048, 256, 0, stream>>>(Wp, wpb, nW);

  // 2) QKV projections, batched over heads via grid.z
  {
    dim3 g((BB * SS) / BM, EE / BN, HH);   // (64, 6, 8)
    gemm_wmma_bf16<false, true, false><<<g, 256, 0, stream>>>(
        xb, wqb, qb, nullptr, EE, EE, EE, EE,
        (size_t)0, (size_t)EE * EE, (size_t)BB * SS * EE);
    gemm_wmma_bf16<false, true, false><<<g, 256, 0, stream>>>(
        xb, wkb, kb, nullptr, EE, EE, EE, EE,
        (size_t)0, (size_t)EE * EE, (size_t)BB * SS * EE);
    gemm_wmma_bf16<false, true, false><<<g, 256, 0, stream>>>(
        xb, wvb, vb, nullptr, EE, EE, EE, EE,
        (size_t)0, (size_t)EE * EE, (size_t)BB * SS * EE);
  }

  // 3) attention per (b,h): scores (NT) -> softmax -> PV (NN)
  const float scale = 0.03608439182435161f;   // 1/sqrt(768)
  dim3 gs(SS / BM, SS / BN, 1);               // (16, 16)
  dim3 gp(SS / BM, EE / BN, 1);               // (16, 6)
  for (int b = 0; b < BB; ++b) {
    for (int h = 0; h < HH; ++h) {
      size_t qo = ((size_t)h * BB * SS + (size_t)b * SS) * EE;
      gemm_wmma_bf16<true, false, false><<<gs, 256, 0, stream>>>(
          qb + qo, kb + qo, att, nullptr, EE, EE, EE, SS,
          (size_t)0, (size_t)0, (size_t)0);
      softmax_rows<<<SS, 256, 0, stream>>>(att, pb, scale);
      gemm_wmma_bf16<false, true, false><<<gp, 256, 0, stream>>>(
          pb, vb + qo, ob + ((size_t)b * SS * HH + h) * EE, nullptr,
          SS, SS, EE, HH * EE,
          (size_t)0, (size_t)0, (size_t)0);
    }
  }

  // 4) output projection with fused bias -> fp32 d_out
  {
    dim3 g((BB * SS) / BM, EE / BN, 1);
    gemm_wmma_bf16<false, false, true><<<g, 256, 0, stream>>>(
        ob, wpb, d_out, bp, HH * EE, HH * EE, EE, EE,
        (size_t)0, (size_t)0, (size_t)0);
  }
  (void)in_sizes; (void)n_in; (void)out_size; (void)ws_size;
}
